// GAT_22617297781315
// MI455X (gfx1250) — matrix operand order repaired and checked by reading, not compiled, not run on hardware
//
#include <hip/hip_runtime.h>
#include <stdint.h>

#define N_NODES   50000
#define N_EDGES   800000
#define N_GRAPHS  64
#define IN_DIM    32
#define HIDDEN    256
#define HEADS     8
#define NEG_SLOPE 0.2f

typedef __bf16 bf16;
typedef __attribute__((ext_vector_type(8)))  __bf16 v8bf;
typedef __attribute__((ext_vector_type(16))) __bf16 v16bf;
typedef __attribute__((ext_vector_type(8)))  float  v8f;

// ---------- helpers ----------
static __device__ __forceinline__ unsigned short f2bf_rne(float f) {
  unsigned int u = __float_as_uint(f);
  u += 0x7FFFu + ((u >> 16) & 1u);   // round-to-nearest-even
  return (unsigned short)(u >> 16);
}

// A-operand fragment: elems 0..15 = K[p..p+15] (contiguous)
static __device__ __forceinline__ v16bf load_afrag(const bf16* p) {
  v8bf lo = *(const v8bf*)p;
  v8bf hi = *(const v8bf*)(p + 8);
  return __builtin_shufflevector(lo, hi, 0,1,2,3,4,5,6,7,8,9,10,11,12,13,14,15);
}
// B-operand fragment: elems 0..15 = K[p..p+15] (contiguous)
static __device__ __forceinline__ v16bf load_bfrag(const bf16* p) {
  v8bf lo = *(const v8bf*)p;
  v8bf hi = *(const v8bf*)(p + 8);
  return __builtin_shufflevector(lo, hi, 0,1,2,3,4,5,6,7,8,9,10,11,12,13,14,15);
}

__global__ void k_f32_to_bf16(const float* __restrict__ in,
                              unsigned short* __restrict__ out, int n) {
  int i = blockIdx.x * blockDim.x + threadIdx.x;
  if (i < n) out[i] = f2bf_rne(in[i]);
}

__global__ void k_iota(int* __restrict__ out, int n) {
  int i = blockIdx.x * blockDim.x + threadIdx.x;
  if (i < n) out[i] = i;
}

// ---------- WMMA GEMM: C[M,N] = A[M,K] * B[N,K]^T  (bf16 in, f32 out) ----------
// One wave32 per 16(M) x 64(N) block: 4 N-tiles share one A fragment.
// M%16==0, N%64==0, K%32==0.
__global__ __launch_bounds__(256)
void k_wmma_gemm(const unsigned short* __restrict__ Au,
                 const unsigned short* __restrict__ Bu,
                 float* __restrict__ C, int M, int N, int K) {
  const bf16* A = (const bf16*)Au;
  const bf16* B = (const bf16*)Bu;
  int wave = (blockIdx.x * blockDim.x + threadIdx.x) >> 5;
  int lane = threadIdx.x & 31;
  int grp_n   = N >> 6;                        // groups of 4 n-tiles
  int ngroups = (M >> 4) * grp_n;
  if (wave >= ngroups) return;
  int m0 = (wave / grp_n) << 4;
  int n0 = (wave % grp_n) << 6;

  // A-operand: lane holds row m0+(lane&15); K base = 16*(lane>=16)
  int akb = (lane >> 4) << 4;
  const bf16* ap = A + (size_t)(m0 + (lane & 15)) * K + akb;
  // B-operand: lane holds column (lane&15); half-wave K split 0..15 / 16..31
  int bkc = (lane >> 4) << 4;
  const bf16* bp = B + (size_t)(n0 + (lane & 15)) * K + bkc;
  const size_t btile = (size_t)16 * K;         // stride between n-tiles

  v8f acc0 = {}, acc1 = {}, acc2 = {}, acc3 = {};
  for (int k0 = 0; k0 < K; k0 += 32) {
    v16bf av = load_afrag(ap + k0);
    v16bf b0 = load_bfrag(bp + k0);
    v16bf b1 = load_bfrag(bp + btile + k0);
    v16bf b2 = load_bfrag(bp + 2 * btile + k0);
    v16bf b3 = load_bfrag(bp + 3 * btile + k0);
    acc0 = __builtin_amdgcn_wmma_f32_16x16x32_bf16(false, av, false, b0, (short)0, acc0, false, false);
    acc1 = __builtin_amdgcn_wmma_f32_16x16x32_bf16(false, av, false, b1, (short)0, acc1, false, false);
    acc2 = __builtin_amdgcn_wmma_f32_16x16x32_bf16(false, av, false, b2, (short)0, acc2, false, false);
    acc3 = __builtin_amdgcn_wmma_f32_16x16x32_bf16(false, av, false, b3, (short)0, acc3, false, false);
  }
  int crow = m0 + ((lane >> 4) << 3);          // M = vgpr + 8*(lane>=16)
  int ccol = n0 + (lane & 15);
  float* cp = C + (size_t)crow * N + ccol;
#pragma unroll
  for (int i = 0; i < 8; i++) {
    cp[(size_t)i * N]      = acc0[i];
    cp[(size_t)i * N + 16] = acc1[i];
    cp[(size_t)i * N + 32] = acc2[i];
    cp[(size_t)i * N + 48] = acc3[i];
  }
}

// ---------- attention logits: al[n,h] = dot(xp[n,h,:], a[h,:]) ----------
__global__ void k_attn_logits(const float* __restrict__ xp,
                              const float* __restrict__ a_src,
                              const float* __restrict__ a_dst,
                              float* __restrict__ al_s, float* __restrict__ al_d,
                              int N, int H, int d) {
  int idx = blockIdx.x * blockDim.x + threadIdx.x;
  if (idx >= N * H) return;
  int n = idx / H, h = idx % H;
  const float* v  = xp + ((size_t)n * H + h) * d;
  const float* as = a_src + (size_t)h * d;
  const float* ad = a_dst + (size_t)h * d;
  float s = 0.f, t = 0.f;
  for (int i = 0; i < d; i++) { float xv = v[i]; s += xv * as[i]; t += xv * ad[i]; }
  al_s[idx] = s; al_d[idx] = t;
}

// ---------- softmax numerators + denominators (segment sums via atomics) ----------
__global__ void k_edge_softmax_num(const int* __restrict__ src, const int* __restrict__ dst,
                                   const float* __restrict__ al_s, const float* __restrict__ al_d,
                                   float* __restrict__ exbuf, float* __restrict__ denom,
                                   int E, int H) {
  int e = blockIdx.x * blockDim.x + threadIdx.x;
  if (e >= E) return;
  int s = src[e], t = dst[e];
#pragma unroll
  for (int h = 0; h < HEADS; h++) {
    float v = al_s[(size_t)s * H + h] + al_d[(size_t)t * H + h];
    v = v > 0.f ? v : v * NEG_SLOPE;             // leaky_relu
    float ex = __expf(v);                        // max-shift skipped (invariant)
    exbuf[(size_t)e * H + h] = ex;
    atomicAdd(&denom[(size_t)t * H + h], ex);
  }
}

// ---------- aggregation, concat layers (head dim = 32): wave per (edge,head) ----------
__global__ void k_edge_agg_concat(const int* __restrict__ src, const int* __restrict__ dst,
                                  const float* __restrict__ exbuf, const float* __restrict__ denom,
                                  const float* __restrict__ xp, float* __restrict__ out,
                                  int E, int H) {
  long gid  = (long)blockIdx.x * blockDim.x + threadIdx.x;
  int  lane = (int)(gid & 31);
  long w    = gid >> 5;
  if (w >= (long)E * H) return;
  int e = (int)(w / H), h = (int)(w % H);
  int s = src[e], t = dst[e];
  float alpha = exbuf[(size_t)e * H + h] / (denom[(size_t)t * H + h] + 1e-16f);
  float v = xp[((size_t)s * H + h) * 32 + lane] * alpha;
  atomicAdd(&out[(size_t)t * (H * 32) + h * 32 + lane], v);
}

// ---------- aggregation, mean layer (head dim = 256, fold 1/H): wave per (edge,head) ----------
__global__ void k_edge_agg_mean(const int* __restrict__ src, const int* __restrict__ dst,
                                const float* __restrict__ exbuf, const float* __restrict__ denom,
                                const float* __restrict__ xp, float* __restrict__ out,
                                int E, int H) {
  long gid  = (long)blockIdx.x * blockDim.x + threadIdx.x;
  int  lane = (int)(gid & 31);
  long w    = gid >> 5;
  if (w >= (long)E * H) return;
  int e = (int)(w / H), h = (int)(w % H);
  int s = src[e], t = dst[e];
  float alpha = exbuf[(size_t)e * H + h] / (denom[(size_t)t * H + h] + 1e-16f);
  alpha *= (1.0f / HEADS);
  const float* xrow = xp + ((size_t)s * H + h) * 256;
  float* orow = out + (size_t)t * 256;
  for (int dd = lane; dd < 256; dd += 32)
    atomicAdd(&orow[dd], alpha * xrow[dd]);
}

// ---------- bias + optional ELU ----------
__global__ void k_bias_act(float* __restrict__ h, const float* __restrict__ b,
                           int N, int C, int do_elu) {
  long idx = (long)blockIdx.x * blockDim.x + threadIdx.x;
  if (idx >= (long)N * C) return;
  float v = h[idx] + b[idx % C];
  if (do_elu) v = v > 0.f ? v : (__expf(v) - 1.0f);
  h[idx] = v;
}

// ---------- global mean pool ----------
__global__ void k_pool_sum(const float* __restrict__ h, const int* __restrict__ batch,
                           float* __restrict__ g, float* __restrict__ cnt, int N, int C) {
  long idx = (long)blockIdx.x * blockDim.x + threadIdx.x;
  if (idx >= (long)N * C) return;
  int n = (int)(idx / C), c = (int)(idx % C);
  int b = batch[n];
  atomicAdd(&g[(size_t)b * C + c], h[idx]);
  if (c == 0) atomicAdd(&cnt[b], 1.0f);
}

__global__ void k_pool_div(float* __restrict__ g, const float* __restrict__ cnt, int G, int C) {
  int idx = blockIdx.x * blockDim.x + threadIdx.x;
  if (idx >= G * C) return;
  g[idx] /= fmaxf(cnt[idx / C], 1.0f);
}

// ---------- tiny dense layer ----------
__global__ void k_linear(const float* __restrict__ in, const float* __restrict__ W,
                         const float* __restrict__ b, float* __restrict__ out,
                         int M, int N, int K, int do_elu) {
  int idx = blockIdx.x * blockDim.x + threadIdx.x;
  if (idx >= M * N) return;
  int m = idx / N, n = idx % N;
  const float* xr = in + (size_t)m * K;
  const float* wr = W + (size_t)n * K;
  float s = b[n];
  for (int k = 0; k < K; k++) s += xr[k] * wr[k];
  if (do_elu) s = s > 0.f ? s : (__expf(s) - 1.0f);
  out[idx] = s;
}

// ========================= launch =========================
extern "C" void kernel_launch(void* const* d_in, const int* in_sizes, int n_in,
                              void* d_out, int out_size, void* d_ws, size_t ws_size,
                              hipStream_t stream) {
  (void)in_sizes; (void)n_in; (void)out_size; (void)ws_size;
  const float* x    = (const float*)d_in[0];
  const int*   ei   = (const int*)d_in[1];
  const int*   batch= (const int*)d_in[2];
  const float* W1   = (const float*)d_in[3];
  const float* as1  = (const float*)d_in[4];
  const float* ad1  = (const float*)d_in[5];
  const float* b1   = (const float*)d_in[6];
  const float* W2   = (const float*)d_in[7];
  const float* as2  = (const float*)d_in[8];
  const float* ad2  = (const float*)d_in[9];
  const float* b2   = (const float*)d_in[10];
  const float* W3   = (const float*)d_in[11];
  const float* as3  = (const float*)d_in[12];
  const float* ad3  = (const float*)d_in[13];
  const float* b3   = (const float*)d_in[14];
  const float* Wc1  = (const float*)d_in[15];
  const float* bc1  = (const float*)d_in[16];
  const float* Wc2  = (const float*)d_in[17];
  const float* bc2  = (const float*)d_in[18];

  const int N = N_NODES, E = N_EDGES, H = HEADS;
  const int* src = ei;
  const int* dst = ei + E;

  // ---- workspace carve-out (~570 MB) ----
  char* p = (char*)d_ws;
  auto alloc = [&](size_t bytes) -> void* {
    void* r = (void*)p; p += (bytes + 255) & ~(size_t)255; return r;
  };
  unsigned short* xbf  = (unsigned short*)alloc((size_t)N * HIDDEN * 2);
  unsigned short* wbf  = (unsigned short*)alloc((size_t)2048 * HIDDEN * 2);
  float* xp     = (float*)alloc((size_t)N * 2048 * 4);
  float* al_s   = (float*)alloc((size_t)N * H * 4);
  float* al_d   = (float*)alloc((size_t)N * H * 4);
  float* exbuf  = (float*)alloc((size_t)E * H * 4);
  float* exself = (float*)alloc((size_t)N * H * 4);
  float* denom  = (float*)alloc((size_t)N * H * 4);
  float* hA     = (float*)alloc((size_t)N * HIDDEN * 4);
  float* hB     = (float*)alloc((size_t)N * HIDDEN * 4);
  int*   iota   = (int*)alloc((size_t)N * 4);
  float* g      = (float*)alloc((size_t)N_GRAPHS * HIDDEN * 4);
  float* cnt    = (float*)alloc((size_t)N_GRAPHS * 4);
  float* z      = (float*)alloc((size_t)N_GRAPHS * 128 * 4);

  auto cdiv = [](long a, long b) { return (unsigned)((a + b - 1) / b); };

  k_iota<<<cdiv(N,256),256,0,stream>>>(iota, N);

  auto gat_layer = [&](const float* hin, int K, const float* W, int OutC,
                       const float* a_s, const float* a_d, const float* bias,
                       float* hout, bool concat, int d, int elu) {
    k_f32_to_bf16<<<cdiv((long)N*K,256),256,0,stream>>>(hin, xbf, N*K);
    k_f32_to_bf16<<<cdiv((long)OutC*K,256),256,0,stream>>>(W, wbf, OutC*K);
    long groups = ((long)N / 16) * ((long)OutC / 64);   // 16x64 per wave
    k_wmma_gemm<<<cdiv(groups*32,256),256,0,stream>>>(xbf, wbf, xp, N, OutC, K);
    k_attn_logits<<<cdiv((long)N*H,256),256,0,stream>>>(xp, a_s, a_d, al_s, al_d, N, H, d);
    hipMemsetAsync(denom, 0, (size_t)N*H*4, stream);
    hipMemsetAsync(hout,  0, (size_t)N*HIDDEN*4, stream);
    k_edge_softmax_num<<<cdiv(E,256),256,0,stream>>>(src, dst, al_s, al_d, exbuf, denom, E, H);
    k_edge_softmax_num<<<cdiv(N,256),256,0,stream>>>(iota, iota, al_s, al_d, exself, denom, N, H);
    if (concat) {
      k_edge_agg_concat<<<cdiv((long)E*H*32,256),256,0,stream>>>(src, dst, exbuf, denom, xp, hout, E, H);
      k_edge_agg_concat<<<cdiv((long)N*H*32,256),256,0,stream>>>(iota, iota, exself, denom, xp, hout, N, H);
    } else {
      k_edge_agg_mean<<<cdiv((long)E*H*32,256),256,0,stream>>>(src, dst, exbuf, denom, xp, hout, E, H);
      k_edge_agg_mean<<<cdiv((long)N*H*32,256),256,0,stream>>>(iota, iota, exself, denom, xp, hout, N, H);
    }
    k_bias_act<<<cdiv((long)N*HIDDEN,256),256,0,stream>>>(hout, bias, N, HIDDEN, elu);
  };

  // layer 1: GATConv(32 -> 8x32, concat) + ELU
  gat_layer(x,  IN_DIM, W1, HIDDEN, as1, ad1, b1, hA, true, 32, 1);
  // layer 2: GATConv(256 -> 8x32, concat) + ELU
  gat_layer(hA, HIDDEN, W2, HIDDEN, as2, ad2, b2, hB, true, 32, 1);
  // layer 3: GATConv(256 -> 8x256, head-mean), no ELU
  gat_layer(hB, HIDDEN, W3, 2048, as3, ad3, b3, hA, false, 256, 0);

  // global mean pool
  hipMemsetAsync(g,   0, (size_t)N_GRAPHS*HIDDEN*4, stream);
  hipMemsetAsync(cnt, 0, (size_t)N_GRAPHS*4, stream);
  k_pool_sum<<<cdiv((long)N*HIDDEN,256),256,0,stream>>>(hA, batch, g, cnt, N, HIDDEN);
  k_pool_div<<<cdiv((long)N_GRAPHS*HIDDEN,256),256,0,stream>>>(g, cnt, N_GRAPHS, HIDDEN);

  // classifier MLP
  k_linear<<<cdiv((long)N_GRAPHS*128,256),256,0,stream>>>(g, Wc1, bc1, z, N_GRAPHS, 128, HIDDEN, 1);
  k_linear<<<1,256,0,stream>>>(z, Wc2, bc2, (float*)d_out, N_GRAPHS, 2, 128, 0);
}
